// HandwritingSynthesisNetwork_40321152975314
// MI455X (gfx1250) — matrix operand order repaired and checked by reading, not compiled, not run on hardware
//
#include <hip/hip_runtime.h>
#include <hip/hip_bf16.h>
#include <cstdint>

// ---------------------------------------------------------------------------
// Problem constants
// ---------------------------------------------------------------------------
#define BB   64
#define TT   800
#define UU   100
#define VV   80
#define HH   512
#define KAA  10
#define NG   2048           // 4*H gate width
#define K0   608            // layer0 GEMM K: 3 + 80 + 13 pad + 512
#define K12  1120           // layer1/2 GEMM K: 595 + 13 pad + 512
#define KFC  1536           // FC K = 3*H
#define NFC  128            // FC N padded (121 real)

// ---------------------------------------------------------------------------
// Workspace layout (bytes, all offsets multiple of 256)
// ---------------------------------------------------------------------------
#define WS_WCAT0   0UL                         // 2048*608  bf16 = 2,490,368
#define WS_WCAT1   2490368UL                   // 2048*1120 bf16 = 4,587,520
#define WS_WCAT2   7077888UL                   // 2048*1120 bf16
#define WS_WFC     11665408UL                  // 128*1536  bf16 =   393,216
#define WS_H0SEQ   12058624UL                  // 51200*512 bf16 = 52,428,800
#define WS_H1SEQ   64487424UL
#define WS_H2SEQ   116916224UL
#define WS_ATTSEQ  169345024UL                 // 51200*80 bf16  =  8,192,000
#define WS_FCOUT   177537024UL                 // 51200*128 f32  = 26,214,400
#define WS_GATES   203751424UL                 // 64*2048 f32    =    524,288
#define WS_H0F32   204275712UL                 // 64*512 f32     =    131,072
#define WS_ZERO    204406784UL                 // zeroed-every-call block:
#define WS_C0      204406784UL                 //   64*512 f32
#define WS_C1      204537856UL
#define WS_C2      204668928UL
#define WS_KAPPA   204800000UL                 //   64*16 f32 (10 used)
#define WS_A0      204804096UL                 //   64*608 bf16
#define WS_A1      204881920UL                 //   64*1120 bf16
#define WS_A2      205025280UL
#define WS_END     205168640UL
#define ZERO_U32   ((WS_END - WS_ZERO) / 4)    // 190,464 dwords

// ---------------------------------------------------------------------------
// WMMA fragment types (CDNA5 wave32, V_WMMA_F32_16X16X32_BF16)
// ---------------------------------------------------------------------------
typedef __attribute__((ext_vector_type(16))) __bf16 v16bf;
typedef __attribute__((ext_vector_type(8)))  float  v8f;

union Frag32B { uint4 u[2]; v16bf v; };

__device__ __forceinline__ float sigf(float x) { return 1.f / (1.f + expf(-x)); }

// ---------------------------------------------------------------------------
// Per-step recurrent GEMM:  out[64,N] = A[64,K](bf16) * W[N,K]^T(bf16) + bias
// One wave per 16x16 tile, 8 waves/block. K multiple of 32, rows 16B aligned.
// ---------------------------------------------------------------------------
__global__ void __launch_bounds__(256) gemm_step(
    const __bf16* __restrict__ A, const __bf16* __restrict__ W,
    const float* __restrict__ bias, float* __restrict__ out,
    int K, int N)
{
  const int lane = threadIdx.x & 31;
  const int gw   = blockIdx.x * 8 + (threadIdx.x >> 5);
  const int mt   = gw & 3;          // 64 rows -> 4 M tiles
  const int nt   = gw >> 2;
  const int sel  = (lane >> 4) & 1; // upper half-wave handles high-K halves
  const int m    = mt * 16 + (lane & 15);
  const int n    = nt * 16 + (lane & 15);
  const __bf16* Arow = A + (size_t)m * K;
  const __bf16* Wrow = W + (size_t)n * K;
  v8f acc = {};
  for (int k0 = 0; k0 < K; k0 += 32) {
    Frag32B fa, fb;
    const int kA = k0 + sel * 8;          // A: K kA..kA+7 and kA+16..kA+23
    fa.u[0] = *(const uint4*)(Arow + kA);
    fa.u[1] = *(const uint4*)(Arow + kA + 16);
    const int kB = k0 + sel * 8;          // B: K kB..kB+7 and kB+16..kB+23 (column n = W row n)
    fb.u[0] = *(const uint4*)(Wrow + kB);
    fb.u[1] = *(const uint4*)(Wrow + kB + 16);
    acc = __builtin_amdgcn_wmma_f32_16x16x32_bf16(false, fa.v, false, fb.v,
                                                  (short)0, acc, false, false);
  }
  const float bn = bias[n];
#pragma unroll
  for (int r = 0; r < 8; ++r) {
    const int mo = mt * 16 + sel * 8 + r; // D layout: VGPR r -> M = r + 8*sel
    out[(size_t)mo * N + n] = acc[r] + bn;
  }
}

// ---------------------------------------------------------------------------
// FC GEMM: A gathered from three [B*T,512] bf16 sequence buffers (K=1536),
// W_fc padded to [128,1536] bf16.  M=51200, 8 N-tiles.
// ---------------------------------------------------------------------------
__global__ void __launch_bounds__(256) gemm_fc(
    const __bf16* __restrict__ S0, const __bf16* __restrict__ S1,
    const __bf16* __restrict__ S2, const __bf16* __restrict__ W,
    const float* __restrict__ bias, float* __restrict__ out)
{
  const int lane = threadIdx.x & 31;
  const int gw   = blockIdx.x * 8 + (threadIdx.x >> 5);
  const int mt   = gw >> 3;
  const int nt   = gw & 7;
  const int sel  = (lane >> 4) & 1;
  const int m    = mt * 16 + (lane & 15);
  const int n    = nt * 16 + (lane & 15);
  const __bf16* Wrow = W + (size_t)n * KFC;
  v8f acc = {};
#pragma unroll
  for (int s = 0; s < 3; ++s) {
    const __bf16* Arow = (s == 0 ? S0 : (s == 1 ? S1 : S2)) + (size_t)m * HH;
    for (int kk = 0; kk < HH; kk += 32) {
      Frag32B fa, fb;
      const int kA = kk + sel * 8;
      fa.u[0] = *(const uint4*)(Arow + kA);
      fa.u[1] = *(const uint4*)(Arow + kA + 16);
      const int kB = s * HH + kk + sel * 8;
      fb.u[0] = *(const uint4*)(Wrow + kB);
      fb.u[1] = *(const uint4*)(Wrow + kB + 16);
      acc = __builtin_amdgcn_wmma_f32_16x16x32_bf16(false, fa.v, false, fb.v,
                                                    (short)0, acc, false, false);
    }
  }
  if (n < 121) {
    const float bn = bias[n];
#pragma unroll
    for (int r = 0; r < 8; ++r) {
      const int mo = mt * 16 + sel * 8 + r;
      out[(size_t)mo * NFC + n] = acc[r] + bn;
    }
  }
}

// ---------------------------------------------------------------------------
// Weight packing: dst[n, 0..Ka) = srcA row, [Ka..padA) = 0, then srcB row.
// ---------------------------------------------------------------------------
__global__ void cat_convert(__bf16* __restrict__ dst, int Kt, int Nreal,
                            const float* __restrict__ srcA, int Ka, int padA,
                            const float* __restrict__ srcB, int Kb, long total)
{
  long idx = (long)blockIdx.x * 256 + threadIdx.x;
  if (idx >= total) return;
  const int n = (int)(idx / Kt), c = (int)(idx % Kt);
  float v = 0.f;
  if (n < Nreal) {
    if (c < Ka)                               v = srcA[(size_t)n * Ka + c];
    else if (srcB && c >= padA && c - padA < Kb) v = srcB[(size_t)n * Kb + (c - padA)];
  }
  dst[idx] = (__bf16)v;
}

__global__ void zero_u32(unsigned* __restrict__ p, int n) {
  const int i = blockIdx.x * 256 + threadIdx.x;
  if (i < n) p[i] = 0u;
}

__global__ void init_a0_x(__bf16* __restrict__ A0, const float* __restrict__ strokes) {
  const int tid = threadIdx.x;
  if (tid < BB * 3) {
    const int b = tid / 3, j = tid % 3;
    A0[(size_t)b * K0 + j] = (__bf16)strokes[(size_t)b * TT * 3 + j];
  }
}

// Pack [x_t | att_t | hprev_t] (cols 0..594) into an A row (K=1120 layers 1/2)
__device__ __forceinline__ void pack_input(
    __bf16* __restrict__ Arow, int c, const float* __restrict__ strokes,
    const __bf16* __restrict__ attseq, const __bf16* __restrict__ prevseq, int bt)
{
  if (c < 3)       Arow[c] = (__bf16)strokes[(size_t)bt * 3 + c];
  else if (c < 83) Arow[c] = attseq[(size_t)bt * VV + (c - 3)];
  else             Arow[c] = prevseq[(size_t)bt * HH + (c - 83)];
}

__global__ void __launch_bounds__(608) pack_init(
    __bf16* __restrict__ A, const float* __restrict__ strokes,
    const __bf16* __restrict__ attseq, const __bf16* __restrict__ prevseq)
{
  const int b = blockIdx.x, c = threadIdx.x;
  if (c < 595) pack_input(A + (size_t)b * K12, c, strokes, attseq, prevseq, b * TT);
}

// ---------------------------------------------------------------------------
// Layer0 pointwise: gates -> (h,c); h to fp32 (for attention), bf16 to A0+seq
// ---------------------------------------------------------------------------
__global__ void __launch_bounds__(256) lstm0_tail(
    const float* __restrict__ gates, float* __restrict__ c_st,
    float* __restrict__ h_f32, __bf16* __restrict__ A0,
    __bf16* __restrict__ h0seq, int t)
{
  const int idx = blockIdx.x * 256 + threadIdx.x;   // 64*512
  const int b = idx >> 9, k = idx & 511;
  const size_t g = (size_t)b * NG + k;
  const float cn = sigf(gates[g + 512]) * c_st[idx] +
                   sigf(gates[g]) * tanhf(gates[g + 1024]);
  const float hn = sigf(gates[g + 1536]) * tanhf(cn);
  c_st[idx]  = cn;
  h_f32[idx] = hn;
  const __bf16 hb = (__bf16)hn;
  A0[(size_t)b * K0 + 96 + k] = hb;
  h0seq[((size_t)b * TT + t) * HH + k] = hb;
}

// ---------------------------------------------------------------------------
// Layers 1/2 pointwise + pack next step's input part. No syncs needed.
// ---------------------------------------------------------------------------
__global__ void __launch_bounds__(608) lstm12_tail(
    const float* __restrict__ gates, float* __restrict__ c_st,
    __bf16* __restrict__ A, __bf16* __restrict__ hseq,
    const float* __restrict__ strokes, const __bf16* __restrict__ attseq,
    const __bf16* __restrict__ prevseq, int t)
{
  const int b = blockIdx.x, tid = threadIdx.x;
  if (tid < HH) {
    const size_t g = (size_t)b * NG + tid;
    const size_t ci = (size_t)b * HH + tid;
    const float cn = sigf(gates[g + 512]) * c_st[ci] +
                     sigf(gates[g]) * tanhf(gates[g + 1024]);
    const float hn = sigf(gates[g + 1536]) * tanhf(cn);
    c_st[ci] = cn;
    const __bf16 hb = (__bf16)hn;
    A[(size_t)b * K12 + 608 + tid] = hb;
    hseq[((size_t)b * TT + t) * HH + tid] = hb;
  }
  if (t + 1 < TT && tid < 595)
    pack_input(A + (size_t)b * K12, tid, strokes, attseq, prevseq, b * TT + t + 1);
}

// ---------------------------------------------------------------------------
// Attention: abk = exp(h0 @ W_att^T + b); kappa += 0.05*exp; phi over U;
// w_t = one-hot contraction via LDS scatter (ds_add_f32). One block per batch.
// Also stages x_{t+1} and w_t (bf16) into A0 for the next step.
// ---------------------------------------------------------------------------
__global__ void __launch_bounds__(128) lstm0_attention(
    const float* __restrict__ h0, const float* __restrict__ W_att,
    const float* __restrict__ b_att, const float* __restrict__ strokes,
    const float* __restrict__ chars_mask, const int* __restrict__ chars,
    float* __restrict__ kappa_st, __bf16* __restrict__ A0,
    __bf16* __restrict__ attseq, int t)
{
  __shared__ float part[120];
  __shared__ float abk[30];
  __shared__ float knew[KAA];
  __shared__ float wacc[VV];
  const int b = blockIdx.x, tid = threadIdx.x;

  if (tid < 120) {                       // 4 threads per output j
    const int j = tid >> 2, q = tid & 3;
    const float* hb = h0 + (size_t)b * HH;
    const float* wr = W_att + (size_t)j * HH;
    float s = 0.f;
    for (int k = q * 128; k < q * 128 + 128; ++k) s += hb[k] * wr[k];
    part[tid] = s;
  }
  if (tid < VV) wacc[tid] = 0.f;
  __syncthreads();
  if (tid < 30)
    abk[tid] = expf(part[tid * 4] + part[tid * 4 + 1] + part[tid * 4 + 2] +
                    part[tid * 4 + 3] + b_att[tid]);
  __syncthreads();
  if (tid < KAA) knew[tid] = abk[20 + tid] * 0.05f + kappa_st[b * KAA + tid];
  __syncthreads();
  if (tid < UU) {
    const float u = (float)tid;
    float phi = 0.f;
#pragma unroll
    for (int j = 0; j < KAA; ++j) {
      const float d = knew[j] - u;
      phi += abk[j] * expf(-abk[10 + j] * d * d);
    }
    phi *= chars_mask[(size_t)b * UU + tid];
    atomicAdd(&wacc[chars[(size_t)b * UU + tid]], phi);
  }
  __syncthreads();
  if (tid < VV) {
    const __bf16 w = (__bf16)wacc[tid];
    attseq[((size_t)b * TT + t) * VV + tid] = w;
    A0[(size_t)b * K0 + 3 + tid] = w;
  }
  if (tid < KAA) kappa_st[b * KAA + tid] = knew[tid];
  if (tid < 3 && t + 1 < TT)
    A0[(size_t)b * K0 + tid] = (__bf16)strokes[((size_t)b * TT + t + 1) * 3 + tid];
}

// ---------------------------------------------------------------------------
// Output head: warp per row. log_softmax(pi), pass-through mu/log_sigma,
// tanh(rho), sigmoid(-eos), packed [log_pi|mu|log_sigma|rho|eos].
// ---------------------------------------------------------------------------
__global__ void __launch_bounds__(256) fc_transform(
    const float* __restrict__ fc, float* __restrict__ out)
{
  const int lane = threadIdx.x & 31;
  const int row  = blockIdx.x * 8 + (threadIdx.x >> 5);   // < 51200
  const float* f = fc + (size_t)row * NFC;
  float pv = (lane < 20) ? f[80 + lane] : -3.402823e38f;
#pragma unroll
  for (int s = 16; s > 0; s >>= 1) pv = fmaxf(pv, __shfl_xor(pv, s, 32));
  float ex = (lane < 20) ? expf(f[80 + lane] - pv) : 0.f;
#pragma unroll
  for (int s = 16; s > 0; s >>= 1) ex += __shfl_xor(ex, s, 32);
  const float lse = pv + logf(ex);
  float* o = out + (size_t)row * 121;
  for (int j = lane; j < 121; j += 32) {
    float v;
    if (j < 20)       v = f[80 + j] - lse;          // log_pi
    else if (j < 100) v = f[j - 20];                // mu (20..59), log_sigma (60..99)
    else if (j < 120) v = tanhf(f[j]);              // rho
    else              v = 1.f / (1.f + expf(f[120])); // sigmoid(-eos)
    o[j] = v;
  }
}

// ---------------------------------------------------------------------------
// Host launch sequence (all on `stream`, graph-capture safe)
// ---------------------------------------------------------------------------
static inline int cdiv_l(long a, long b) { return (int)((a + b - 1) / b); }

extern "C" void kernel_launch(void* const* d_in, const int* in_sizes, int n_in,
                              void* d_out, int out_size, void* d_ws, size_t ws_size,
                              hipStream_t stream) {
  (void)in_sizes; (void)n_in; (void)out_size; (void)ws_size;
  const int*   chars      = (const int*)d_in[0];
  const float* chars_mask = (const float*)d_in[1];
  const float* strokes    = (const float*)d_in[2];
  /* d_in[3] strokes_mask unused by the math */
  const float* W_ih0 = (const float*)d_in[4];
  const float* W_hh0 = (const float*)d_in[5];
  const float* b0    = (const float*)d_in[6];
  const float* W_att = (const float*)d_in[7];
  const float* b_att = (const float*)d_in[8];
  const float* W_ih1 = (const float*)d_in[9];
  const float* W_hh1 = (const float*)d_in[10];
  const float* b1    = (const float*)d_in[11];
  const float* W_ih2 = (const float*)d_in[12];
  const float* W_hh2 = (const float*)d_in[13];
  const float* b2    = (const float*)d_in[14];
  const float* W_fc  = (const float*)d_in[15];
  const float* b_fc  = (const float*)d_in[16];
  float* out = (float*)d_out;

  char* ws = (char*)d_ws;
  __bf16* Wc0    = (__bf16*)(ws + WS_WCAT0);
  __bf16* Wc1    = (__bf16*)(ws + WS_WCAT1);
  __bf16* Wc2    = (__bf16*)(ws + WS_WCAT2);
  __bf16* Wfc    = (__bf16*)(ws + WS_WFC);
  __bf16* h0seq  = (__bf16*)(ws + WS_H0SEQ);
  __bf16* h1seq  = (__bf16*)(ws + WS_H1SEQ);
  __bf16* h2seq  = (__bf16*)(ws + WS_H2SEQ);
  __bf16* attseq = (__bf16*)(ws + WS_ATTSEQ);
  float*  fcout  = (float*)(ws + WS_FCOUT);
  float*  gates  = (float*)(ws + WS_GATES);
  float*  h0f    = (float*)(ws + WS_H0F32);
  float*  c0     = (float*)(ws + WS_C0);
  float*  c1     = (float*)(ws + WS_C1);
  float*  c2     = (float*)(ws + WS_C2);
  float*  kappa  = (float*)(ws + WS_KAPPA);
  __bf16* A0     = (__bf16*)(ws + WS_A0);
  __bf16* A1     = (__bf16*)(ws + WS_A1);
  __bf16* A2     = (__bf16*)(ws + WS_A2);

  // --- one-time (per call) packing / state init ---
  {
    long t0 = (long)NG * K0;
    cat_convert<<<cdiv_l(t0, 256), 256, 0, stream>>>(Wc0, K0, NG, W_ih0, 83, 96, W_hh0, HH, t0);
    long t1 = (long)NG * K12;
    cat_convert<<<cdiv_l(t1, 256), 256, 0, stream>>>(Wc1, K12, NG, W_ih1, 595, 608, W_hh1, HH, t1);
    cat_convert<<<cdiv_l(t1, 256), 256, 0, stream>>>(Wc2, K12, NG, W_ih2, 595, 608, W_hh2, HH, t1);
    long tf = (long)NFC * KFC;
    cat_convert<<<cdiv_l(tf, 256), 256, 0, stream>>>(Wfc, KFC, 121, W_fc, KFC, KFC, nullptr, 0, tf);
  }
  zero_u32<<<cdiv_l(ZERO_U32, 256), 256, 0, stream>>>((unsigned*)(ws + WS_ZERO), (int)ZERO_U32);
  init_a0_x<<<1, 192, 0, stream>>>(A0, strokes);

  // --- layer 0 + attention (sequential over T) ---
  for (int t = 0; t < TT; ++t) {
    gemm_step<<<64, 256, 0, stream>>>(A0, Wc0, b0, gates, K0, NG);       // 512 tiles
    lstm0_tail<<<128, 256, 0, stream>>>(gates, c0, h0f, A0, h0seq, t);
    lstm0_attention<<<BB, 128, 0, stream>>>(h0f, W_att, b_att, strokes,
                                            chars_mask, chars, kappa, A0, attseq, t);
  }

  // --- layer 1 ---
  pack_init<<<BB, 608, 0, stream>>>(A1, strokes, attseq, h0seq);
  for (int t = 0; t < TT; ++t) {
    gemm_step<<<64, 256, 0, stream>>>(A1, Wc1, b1, gates, K12, NG);
    lstm12_tail<<<BB, 608, 0, stream>>>(gates, c1, A1, h1seq, strokes, attseq, h0seq, t);
  }

  // --- layer 2 ---
  pack_init<<<BB, 608, 0, stream>>>(A2, strokes, attseq, h1seq);
  for (int t = 0; t < TT; ++t) {
    gemm_step<<<64, 256, 0, stream>>>(A2, Wc2, b2, gates, K12, NG);
    lstm12_tail<<<BB, 608, 0, stream>>>(gates, c2, A2, h2seq, strokes, attseq, h1seq, t);
  }

  // --- FC head + output transforms ---
  gemm_fc<<<(BB * TT / 16) * 8 / 8, 256, 0, stream>>>(h0seq, h1seq, h2seq, Wfc, b_fc, fcout);
  fc_transform<<<BB * TT / 8, 256, 0, stream>>>(fcout, out);
}
